// Encoder_28106265985101
// MI455X (gfx1250) — matrix operand order repaired and checked by reading, not compiled, not run on hardware
//
#include <hip/hip_runtime.h>

#define Bb 32
#define Ss 512
#define Hh 512
#define Ll 6
#define NHh 8
#define FFf 2048
#define HDd 64
#define Jj 45
#define NOPp 4
#define NDOMd 5

typedef __attribute__((ext_vector_type(16))) __bf16 bf16x16;
typedef __attribute__((ext_vector_type(8)))  float  floatx8;

union FragAB { bf16x16 v; uint4 q[2]; };

__device__ __forceinline__ unsigned short f2bf(float f) {
  unsigned u = __float_as_uint(f);
  u += 0x7FFFu + ((u >> 16) & 1u);   // round-to-nearest-even
  return (unsigned short)(u >> 16);
}

__device__ __forceinline__ floatx8 wmma_bf16(FragAB a, FragAB b, floatx8 c) {
  return __builtin_amdgcn_wmma_f32_16x16x32_bf16(false, a.v, false, b.v,
                                                 (short)0, c, false, false);
}

__device__ __forceinline__ float gelu_tanh(float x) {
  float x3 = x * x * x;
  return 0.5f * x * (1.0f + tanhf(0.7978845608028654f * (x + 0.044715f * x3)));
}

// async global->LDS copy of one 16-byte chunk per lane (CDNA5 TDM-lite path)
__device__ __forceinline__ void async_cp16(const void* gaddr, void* lds) {
  unsigned l32 = (unsigned)(unsigned long long)lds;
  asm volatile("global_load_async_to_lds_b128 %0, %1, off"
               :: "v"(l32), "v"((unsigned long long)gaddr) : "memory");
}
__device__ __forceinline__ void async_wait0() {
  asm volatile("s_wait_asynccnt 0x0" ::: "memory");
}

// out[N,K] (bf16) = transpose(in[K,N] f32)
__global__ __launch_bounds__(256)
void wconv_t(const float* __restrict__ in, unsigned short* __restrict__ out,
             int K, int N) {
  int idx = blockIdx.x * 256 + threadIdx.x;
  if (idx >= K * N) return;
  int n = idx / K, k = idx % K;
  out[idx] = f2bf(in[(size_t)k * N + n]);
}

// C[M,N] = A[M,K](bf16) * Bt[N,K](bf16)^T + bias
// Block tile 128x128, BK=32, double-buffered async LDS staging.
// 8 waves: wr=w&3 -> 32 rows, wc=w>>2 -> 64 cols; 8 WMMAs / wave / k-step.
// EPI: 0=f32 1=bf16 2=gelu->bf16 3=tanh->f32
template <int EPI>
__global__ __launch_bounds__(256)
void gemm_bf16(const unsigned short* __restrict__ A,
               const unsigned short* __restrict__ Bt,
               const float* __restrict__ bias,
               float* __restrict__ Cf, unsigned short* __restrict__ Cb,
               int M, int N, int K) {
  __shared__ __align__(16) unsigned short tA[2][128][32];
  __shared__ __align__(16) unsigned short tB[2][128][32];
  const int tid  = threadIdx.x;
  const int lane = tid & 31;
  const int w    = tid >> 5;
  const int wr = w & 3, wc = w >> 2;
  const int m0 = blockIdx.y * 128;
  const int n0 = blockIdx.x * 128;
  const int lm  = lane & 15;
  const int kbA = (lane < 16) ? 0 : 8;    // A-frag K base (ISA 16-bit A layout)
  const int khB = (lane < 16) ? 0 : 8;    // B-frag K base (same k order as A)
  const int nk = K >> 5;

  auto stage = [&](int ks, int buf) {
    const int k0 = ks * 32;
#pragma unroll
    for (int i = 0; i < 2; ++i) {         // 512 chunks of 16B per tile pair half
      int c = tid + 256 * i;
      int r = c >> 2, ko = (c & 3) * 8;
      int ar = m0 + r; if (ar >= M) ar = M - 1;
      async_cp16(A  + (size_t)ar * K + k0 + ko,        &tA[buf][r][ko]);
      async_cp16(Bt + (size_t)(n0 + r) * K + k0 + ko,  &tB[buf][r][ko]);
    }
  };

  floatx8 acc[2][4] = {};
  stage(0, 0);
  for (int ks = 0; ks < nk; ++ks) {
    async_wait0();          // my async writes for tile `ks` are in LDS
    __syncthreads();        // everyone's are; prev buffer fully consumed
    if (ks + 1 < nk) stage(ks + 1, (ks + 1) & 1);
    const int buf = ks & 1;
    FragAB fa[2];
#pragma unroll
    for (int s = 0; s < 2; ++s) {
      const unsigned short* ap = &tA[buf][wr * 32 + s * 16 + lm][0];
      fa[s].q[0] = *(const uint4*)(ap + kbA);
      fa[s].q[1] = *(const uint4*)(ap + 16 + kbA);
    }
#pragma unroll
    for (int t = 0; t < 4; ++t) {
      const unsigned short* bp = &tB[buf][wc * 64 + t * 16 + lm][khB];
      FragAB fb;
      fb.q[0] = ((const uint4*)bp)[0];
      fb.q[1] = *(const uint4*)(bp + 16);
      acc[0][t] = wmma_bf16(fa[0], fb, acc[0][t]);
      acc[1][t] = wmma_bf16(fa[1], fb, acc[1][t]);
    }
  }

  const int rhalf = (lane < 16) ? 0 : 8;
#pragma unroll
  for (int s = 0; s < 2; ++s) {
    const int rbase = m0 + wr * 32 + s * 16 + rhalf;
#pragma unroll
    for (int t = 0; t < 4; ++t) {
      const int col = n0 + wc * 64 + t * 16 + lm;
      const float bvv = bias ? bias[col] : 0.0f;
#pragma unroll
      for (int i = 0; i < 8; ++i) {
        const int row = rbase + i;
        if (row >= M) continue;
        float vv = acc[s][t][i] + bvv;
        size_t o = (size_t)row * N + col;
        if (EPI == 0)      Cf[o] = vv;
        else if (EPI == 1) Cb[o] = f2bf(vv);
        else if (EPI == 2) Cb[o] = f2bf(gelu_tanh(vv));
        else               Cf[o] = tanhf(vv);
      }
    }
  }
}

// Flash attention: 1 wave = 16 q-rows x full HD=64, key blocks of 32.
__global__ __launch_bounds__(128)
void flash_attn(const unsigned short* __restrict__ q,
                const unsigned short* __restrict__ k,
                const unsigned short* __restrict__ v,
                const int* __restrict__ amask,
                unsigned short* __restrict__ ctx) {
  __shared__ __align__(16) unsigned short vt[4][HDd][32]; // V^T stage per wave
  __shared__ __align__(16) unsigned short pl[4][16][32];  // P stage per wave
  const int lane = threadIdx.x & 31;
  const int w    = threadIdx.x >> 5;
  const int qg = blockIdx.x % (Ss / 64);
  const int bh = blockIdx.x / (Ss / 64);
  const int b = bh / NHh, hh = bh % NHh;
  const int q0 = qg * 64 + w * 16;
  const int lm  = lane & 15;
  const int kbA = (lane < 16) ? 0 : 8;
  const int khB = (lane < 16) ? 0 : 8;
  const size_t base = ((size_t)b * Ss) * Hh + (size_t)hh * HDd;

  FragAB fq[2];
  {
    const unsigned short* qp = q + base + (size_t)(q0 + lm) * Hh;
#pragma unroll
    for (int d = 0; d < 2; ++d) {
      fq[d].q[0] = *(const uint4*)(qp + d * 32 + kbA);
      fq[d].q[1] = *(const uint4*)(qp + d * 32 + 16 + kbA);
    }
  }
  floatx8 acc[4] = {};
  float mrow[8], lrow[8];
#pragma unroll
  for (int i = 0; i < 8; ++i) { mrow[i] = -1e30f; lrow[i] = 0.0f; }

  for (int kb = 0; kb < Ss; kb += 32) {
    { // stage V block transposed into LDS: vt[d][key]
      const unsigned short* vp = v + base + (size_t)(kb + lane) * Hh;
#pragma unroll 8
      for (int d = 0; d < HDd; ++d) vt[w][d][lane] = vp[d];
    }
    floatx8 s2[2] = {};
#pragma unroll
    for (int t = 0; t < 2; ++t) {
      const unsigned short* kp = k + base + (size_t)(kb + t * 16 + lm) * Hh;
#pragma unroll
      for (int d = 0; d < 2; ++d) {
        FragAB fk;
        fk.q[0] = ((const uint4*)(kp + d * 32 + khB))[0];
        fk.q[1] = *(const uint4*)(kp + d * 32 + 16 + khB);
        s2[t] = wmma_bf16(fq[d], fk, s2[t]);
      }
      const int key = kb + t * 16 + lm;
      const float mb = (1.0f - (float)amask[b * Ss + key]) * -10000.0f;
#pragma unroll
      for (int i = 0; i < 8; ++i) s2[t][i] = s2[t][i] * 0.125f + mb;
    }
    // online softmax (rows split: lanes0-15 row i, lanes16-31 row i+8)
#pragma unroll
    for (int i = 0; i < 8; ++i) {
      float mx = fmaxf(s2[0][i], s2[1][i]);
      for (int off = 1; off < 16; off <<= 1) mx = fmaxf(mx, __shfl_xor(mx, off, 32));
      const float mnew = fmaxf(mrow[i], mx);
      const float al = __expf(mrow[i] - mnew);
      const float p0 = __expf(s2[0][i] - mnew);
      const float p1 = __expf(s2[1][i] - mnew);
      s2[0][i] = p0; s2[1][i] = p1;
      float sm = p0 + p1;
      for (int off = 1; off < 16; off <<= 1) sm += __shfl_xor(sm, off, 32);
      lrow[i] = lrow[i] * al + sm;
      mrow[i] = mnew;
#pragma unroll
      for (int c = 0; c < 4; ++c) acc[c][i] *= al;
    }
    // P (C-frag layout) -> LDS -> A-frag layout
#pragma unroll
    for (int i = 0; i < 8; ++i) {
      const int r = i + ((lane < 16) ? 0 : 8);
      pl[w][r][lm]      = f2bf(s2[0][i]);
      pl[w][r][16 + lm] = f2bf(s2[1][i]);
    }
    FragAB fp;
    fp.q[0] = *(const uint4*)(&pl[w][lm][kbA]);
    fp.q[1] = *(const uint4*)(&pl[w][lm][16 + kbA]);
#pragma unroll
    for (int c = 0; c < 4; ++c) {
      FragAB fv;
      const unsigned short* vr = &vt[w][c * 16 + lm][khB];
      fv.q[0] = ((const uint4*)vr)[0];
      fv.q[1] = *(const uint4*)(vr + 16);
      acc[c] = wmma_bf16(fp, fv, acc[c]);
    }
  }
#pragma unroll
  for (int c = 0; c < 4; ++c)
#pragma unroll
    for (int i = 0; i < 8; ++i) {
      const int row = q0 + i + ((lane < 16) ? 0 : 8);
      ctx[base + (size_t)row * Hh + c * 16 + lm] = f2bf(acc[c][i] / lrow[i]);
    }
}

__global__ __launch_bounds__(128)
void embed_ln(const int* __restrict__ ids, const int* __restrict__ tts,
              const float* __restrict__ tok, const float* __restrict__ pos,
              const float* __restrict__ typ, const float* __restrict__ g,
              const float* __restrict__ bet, float* __restrict__ outf,
              unsigned short* __restrict__ outb) {
  const int lane = threadIdx.x & 31;
  const int row  = blockIdx.x * 4 + (threadIdx.x >> 5);
  const int s = row % Ss;
  const int id = ids[row], ty = tts[row];
  float vals[16]; float sum = 0.f;
#pragma unroll
  for (int i = 0; i < 16; ++i) {
    int c = lane + i * 32;
    float x = tok[(size_t)id * Hh + c] + pos[(size_t)s * Hh + c] + typ[(size_t)ty * Hh + c];
    vals[i] = x; sum += x;
  }
  for (int off = 16; off; off >>= 1) sum += __shfl_xor(sum, off, 32);
  const float mean = sum * (1.0f / Hh);
  float var = 0.f;
#pragma unroll
  for (int i = 0; i < 16; ++i) { float d = vals[i] - mean; var += d * d; }
  for (int off = 16; off; off >>= 1) var += __shfl_xor(var, off, 32);
  const float r = rsqrtf(var * (1.0f / Hh) + 1e-12f);
  const size_t ro = (size_t)row * Hh;
#pragma unroll
  for (int i = 0; i < 16; ++i) {
    int c = lane + i * 32;
    float o = g[c] * (vals[i] - mean) * r + bet[c];
    outf[ro + c] = o;
    outb[ro + c] = f2bf(o);
  }
}

__global__ __launch_bounds__(128)
void resid_ln(const float* __restrict__ x, const float* __restrict__ d,
              const float* __restrict__ g, const float* __restrict__ bet,
              float* __restrict__ outf, unsigned short* __restrict__ outb) {
  const int lane = threadIdx.x & 31;
  const int row  = blockIdx.x * 4 + (threadIdx.x >> 5);
  const size_t ro = (size_t)row * Hh;
  float vals[16]; float sum = 0.f;
#pragma unroll
  for (int i = 0; i < 16; ++i) {
    int c = lane + i * 32;
    float v = x[ro + c] + d[ro + c];
    vals[i] = v; sum += v;
  }
  for (int off = 16; off; off >>= 1) sum += __shfl_xor(sum, off, 32);
  const float mean = sum * (1.0f / Hh);
  float var = 0.f;
#pragma unroll
  for (int i = 0; i < 16; ++i) { float dd = vals[i] - mean; var += dd * dd; }
  for (int off = 16; off; off >>= 1) var += __shfl_xor(var, off, 32);
  const float r = rsqrtf(var * (1.0f / Hh) + 1e-12f);
#pragma unroll
  for (int i = 0; i < 16; ++i) {
    int c = lane + i * 32;
    float o = g[c] * (vals[i] - mean) * r + bet[c];
    outf[ro + c] = o;
    outb[ro + c] = f2bf(o);
  }
}

__global__ __launch_bounds__(256)
void cls_gather(const unsigned short* __restrict__ hbf, unsigned short* __restrict__ cls) {
  int idx = blockIdx.x * 256 + threadIdx.x;
  if (idx >= Bb * Hh) return;
  int b = idx / Hh, c = idx % Hh;
  cls[idx] = hbf[((size_t)b * Ss) * Hh + c];
}

__global__ __launch_bounds__(256)
void state_gather(const float* __restrict__ seq, const int* __restrict__ pos,
                  float* __restrict__ st) {
  int idx = blockIdx.x * 256 + threadIdx.x;
  if (idx >= Bb * Jj * Hh) return;
  int c = idx % Hh; int bj = idx / Hh; int b = bj / Jj, j = bj % Jj;
  int p = pos[b * Jj + j];
  st[idx] = seq[((size_t)b * Ss + p) * Hh + c];
}

// one wave per (row, out-col) dot product over H
__global__ __launch_bounds__(256)
void head_scores(const float* __restrict__ xin, const float* __restrict__ W,
                 const float* __restrict__ bias, float* __restrict__ out,
                 int rows, int nout) {
  int id = (blockIdx.x * 256 + threadIdx.x) >> 5;
  int lane = threadIdx.x & 31;
  if (id >= rows * nout) return;
  int rrow = id / nout, o = id % nout;
  const float* xp = xin + (size_t)rrow * Hh;
  float sum = 0.f;
#pragma unroll
  for (int i = 0; i < 16; ++i) { int c = lane + i * 32; sum += xp[c] * W[(size_t)c * nout + o]; }
  for (int off = 16; off; off >>= 1) sum += __shfl_xor(sum, off, 32);
  if (lane == 0) out[(size_t)rrow * nout + o] = sum + bias[o];
}

// stable partition of op_ids==0 first, pad with zeros (masked_select + pad)
__global__ __launch_bounds__(128)
void decoder_gather(const float* __restrict__ st, const int* __restrict__ op_ids,
                    const int* __restrict__ maxu, float* __restrict__ out) {
  __shared__ int order[Jj];
  __shared__ int cnt_s;
  int b = blockIdx.x;
  if (threadIdx.x == 0) {
    int c = 0;
    for (int j = 0; j < Jj; ++j) if (op_ids[b * Jj + j] == 0) order[c++] = j;
    cnt_s = c;
    for (int j = 0; j < Jj; ++j) if (op_ids[b * Jj + j] != 0) order[c++] = j;
  }
  __syncthreads();
  int mu = maxu[0]; if (mu > Jj) mu = Jj;
  int cnt = cnt_s;
  for (int j = 0; j < mu; ++j) {
    float sc = (j < cnt) ? 1.f : 0.f;
    int src = order[j];
    for (int t = threadIdx.x; t < Hh; t += blockDim.x)
      out[((size_t)b * Jj + j) * Hh + t] = st[((size_t)b * Jj + src) * Hh + t] * sc;
  }
  for (int j = mu; j < Jj; ++j)
    for (int t = threadIdx.x; t < Hh; t += blockDim.x)
      out[((size_t)b * Jj + j) * Hh + t] = 0.f;
}

extern "C" void kernel_launch(void* const* d_in, const int* in_sizes, int n_in,
                              void* d_out, int out_size, void* d_ws, size_t ws_size,
                              hipStream_t stream) {
  (void)in_sizes; (void)n_in; (void)out_size; (void)ws_size;
  const int*   input_ids  = (const int*)d_in[0];
  const int*   token_type = (const int*)d_in[1];
  const int*   state_pos  = (const int*)d_in[2];
  const int*   amask      = (const int*)d_in[3];
  const int*   op_ids     = (const int*)d_in[4];
  const int*   max_upd    = (const int*)d_in[5];
  const float* tok_emb = (const float*)d_in[6];
  const float* pos_emb = (const float*)d_in[7];
  const float* type_emb = (const float*)d_in[8];
  const float* eg = (const float*)d_in[9];
  const float* eb = (const float*)d_in[10];
  const float* Wq = (const float*)d_in[11]; const float* bq = (const float*)d_in[12];
  const float* Wk = (const float*)d_in[13]; const float* bk = (const float*)d_in[14];
  const float* Wv = (const float*)d_in[15]; const float* bv = (const float*)d_in[16];
  const float* Wo = (const float*)d_in[17]; const float* bo = (const float*)d_in[18];
  const float* ln1g = (const float*)d_in[19]; const float* ln1b = (const float*)d_in[20];
  const float* W1 = (const float*)d_in[21]; const float* b1 = (const float*)d_in[22];
  const float* W2 = (const float*)d_in[23]; const float* b2 = (const float*)d_in[24];
  const float* ln2g = (const float*)d_in[25]; const float* ln2b = (const float*)d_in[26];
  const float* poolW = (const float*)d_in[27]; const float* poolB = (const float*)d_in[28];
  const float* actW = (const float*)d_in[29]; const float* actB = (const float*)d_in[30];
  const float* domW = (const float*)d_in[31]; const float* domB = (const float*)d_in[32];

  float* out = (float*)d_out;
  const size_t DOM_OFF  = 0;
  const size_t SS_OFF   = DOM_OFF + (size_t)Bb * NDOMd;
  const size_t DEC_OFF  = SS_OFF  + (size_t)Bb * Jj * NOPp;
  const size_t SEQ_OFF  = DEC_OFF + (size_t)Bb * Jj * Hh;
  const size_t POOL_OFF = SEQ_OFF + (size_t)Bb * Ss * Hh;

  char* ws = (char*)d_ws;
  size_t off = 0;
  auto take = [&](size_t bytes) -> void* {
    off = (off + 255) & ~(size_t)255;
    void* p = ws + off;
    off += bytes;
    return p;
  };
  const size_t TOK = (size_t)Bb * Ss;
  const size_t LW  = 4 * (size_t)Hh * Hh + 2 * (size_t)Hh * FFf; // per-layer bf16 weight elems
  unsigned short* wt   = (unsigned short*)take(((size_t)Ll * LW + (size_t)Hh * Hh) * 2);
  float*          h    = (float*)take(TOK * Hh * 4);
  unsigned short* hbf  = (unsigned short*)take(TOK * Hh * 2);
  unsigned short* qbf  = (unsigned short*)take(TOK * Hh * 2);
  unsigned short* kbf  = (unsigned short*)take(TOK * Hh * 2);
  unsigned short* vbf  = (unsigned short*)take(TOK * Hh * 2);
  unsigned short* cbf  = (unsigned short*)take(TOK * Hh * 2);
  float*          tmp  = (float*)take(TOK * Hh * 4);
  unsigned short* ffbf = (unsigned short*)take(TOK * FFf * 2);
  unsigned short* clsb = (unsigned short*)take((size_t)Bb * Hh * 2);
  float*          st   = (float*)take((size_t)Bb * Jj * Hh * 4);

  const int nHH = Hh * Hh, nHF = Hh * FFf;
  for (int l = 0; l < Ll; ++l) {
    unsigned short* p = wt + (size_t)l * LW;
    wconv_t<<<dim3((nHH + 255) / 256), 256, 0, stream>>>(Wq + (size_t)l * nHH, p,                    Hh, Hh);
    wconv_t<<<dim3((nHH + 255) / 256), 256, 0, stream>>>(Wk + (size_t)l * nHH, p + (size_t)nHH,      Hh, Hh);
    wconv_t<<<dim3((nHH + 255) / 256), 256, 0, stream>>>(Wv + (size_t)l * nHH, p + 2 * (size_t)nHH,  Hh, Hh);
    wconv_t<<<dim3((nHH + 255) / 256), 256, 0, stream>>>(Wo + (size_t)l * nHH, p + 3 * (size_t)nHH,  Hh, Hh);
    wconv_t<<<dim3((nHF + 255) / 256), 256, 0, stream>>>(W1 + (size_t)l * nHF, p + 4 * (size_t)nHH,  Hh, FFf);
    wconv_t<<<dim3((nHF + 255) / 256), 256, 0, stream>>>(W2 + (size_t)l * nHF, p + 4 * (size_t)nHH + (size_t)nHF, FFf, Hh);
  }
  unsigned short* poolT = wt + (size_t)Ll * LW;
  wconv_t<<<dim3((nHH + 255) / 256), 256, 0, stream>>>(poolW, poolT, Hh, Hh);

  embed_ln<<<dim3((unsigned)(TOK / 4)), dim3(128), 0, stream>>>(
      input_ids, token_type, tok_emb, pos_emb, type_emb, eg, eb, h, hbf);

  float* seq = out + SEQ_OFF;
  dim3 gridH(Hh / 128, (unsigned)(TOK / 128));
  dim3 gridF(FFf / 128, (unsigned)(TOK / 128));
  for (int l = 0; l < Ll; ++l) {
    unsigned short* p = wt + (size_t)l * LW;
    gemm_bf16<1><<<gridH, 256, 0, stream>>>(hbf, p,                    bq + (size_t)l * Hh, nullptr, qbf, (int)TOK, Hh, Hh);
    gemm_bf16<1><<<gridH, 256, 0, stream>>>(hbf, p + (size_t)nHH,      bk + (size_t)l * Hh, nullptr, kbf, (int)TOK, Hh, Hh);
    gemm_bf16<1><<<gridH, 256, 0, stream>>>(hbf, p + 2 * (size_t)nHH,  bv + (size_t)l * Hh, nullptr, vbf, (int)TOK, Hh, Hh);
    flash_attn<<<dim3(Bb * NHh * (Ss / 64)), dim3(128), 0, stream>>>(qbf, kbf, vbf, amask, cbf);
    gemm_bf16<0><<<gridH, 256, 0, stream>>>(cbf, p + 3 * (size_t)nHH,  bo + (size_t)l * Hh, tmp, nullptr, (int)TOK, Hh, Hh);
    resid_ln<<<dim3((unsigned)(TOK / 4)), dim3(128), 0, stream>>>(
        h, tmp, ln1g + (size_t)l * Hh, ln1b + (size_t)l * Hh, h, hbf);
    gemm_bf16<2><<<gridF, 256, 0, stream>>>(hbf, p + 4 * (size_t)nHH,  b1 + (size_t)l * FFf, nullptr, ffbf, (int)TOK, FFf, Hh);
    gemm_bf16<0><<<gridH, 256, 0, stream>>>(ffbf, p + 4 * (size_t)nHH + (size_t)nHF, b2 + (size_t)l * Hh, tmp, nullptr, (int)TOK, Hh, FFf);
    float* lnout = (l == Ll - 1) ? seq : h;
    resid_ln<<<dim3((unsigned)(TOK / 4)), dim3(128), 0, stream>>>(
        h, tmp, ln2g + (size_t)l * Hh, ln2b + (size_t)l * Hh, lnout, hbf);
  }

  cls_gather<<<dim3((Bb * Hh + 255) / 256), 256, 0, stream>>>(hbf, clsb);
  gemm_bf16<3><<<dim3(Hh / 128, 1), 256, 0, stream>>>(clsb, poolT, poolB, out + POOL_OFF, nullptr, Bb, Hh, Hh);
  state_gather<<<dim3((Bb * Jj * Hh + 255) / 256), 256, 0, stream>>>(seq, state_pos, st);
  head_scores<<<dim3((Bb * Jj * NOPp * 32 + 255) / 256), 256, 0, stream>>>(st, actW, actB, out + SS_OFF, Bb * Jj, NOPp);
  head_scores<<<dim3((Bb * NDOMd * 32 + 255) / 256), 256, 0, stream>>>(out + POOL_OFF, domW, domB, out + DOM_OFF, Bb, NDOMd);
  decoder_gather<<<dim3(Bb), dim3(128), 0, stream>>>(st, op_ids, max_upd, out + DEC_OFF);
}
